// Attention_18932215841487
// MI455X (gfx1250) — matrix operand order repaired and checked by reading, not compiled, not run on hardware
//
#include <hip/hip_runtime.h>
#include <hip/hip_bf16.h>

typedef __attribute__((ext_vector_type(16))) _Float16 v16h;
typedef __attribute__((ext_vector_type(8)))  _Float16 v8h;
typedef __attribute__((ext_vector_type(8)))  float    v8f;

#define B_       2
#define S_       2048
#define DIM_     4096
#define NH_      32
#define NKV_     8
#define HD_      128
#define THETA_   500000.0f

// ---------- WMMA fragment loaders (16-bit, contiguous k order, wave32) ----------

// A-matrix 16x32 f16: lane l holds row m=l&15; elem i -> K=kh+i,
// kh = 16*(l>>4).  Source row-major [m][k].
__device__ __forceinline__ v16h load_A_frag(const _Float16* base, int row_stride, int lane) {
    int m  = lane & 15;
    int kh = (lane >> 4) << 4;
    const _Float16* p = base + (size_t)m * row_stride + kh;
    v16h a;
#pragma unroll
    for (int i = 0; i < 8; ++i) a[i] = p[i];
#pragma unroll
    for (int i = 0; i < 8; ++i) a[8 + i] = p[8 + i];
    return a;
}

// B-matrix 32x16 f16: lane l holds col n=l&15; elem i -> K=(l>>4)*16+i.
// Source stored K-transposed: [n][k], so each lane reads 16 contiguous halfs.
__device__ __forceinline__ v16h load_B_fragT(const _Float16* base, int row_stride, int lane) {
    int n  = lane & 15;
    int kk = (lane >> 4) << 4;
    const _Float16* p = base + (size_t)n * row_stride + kk;
    v16h b;
#pragma unroll
    for (int i = 0; i < 16; ++i) b[i] = p[i];
    return b;
}

// ---------- CDNA5 async global->LDS copy (16 bytes per lane) ----------
// Generic LDS pointer: low 32 bits are the LDS byte offset (aperture layout).
__device__ __forceinline__ void async_copy_b128(const _Float16* gsrc, _Float16* ldst) {
    uint32_t loff = (uint32_t)(uintptr_t)ldst;
    asm volatile("global_load_async_to_lds_b128 %0, %1, off"
                 :: "v"(loff), "v"(gsrc)
                 : "memory");
}
__device__ __forceinline__ void wait_asynccnt0() {
    asm volatile("s_wait_asynccnt 0x0" ::: "memory");
}

// ---------- Generic WMMA GEMM: C[MxN] = A[MxK] * Bw[KxN] ----------
// Double-buffered LDS, global loads for tile k+1 overlapped with WMMA on tile k.
// A_HALF: A is f16 (else f32, converted while staging). OUT_FLOAT: store f32 else f16.
template <bool A_HALF, bool OUT_FLOAT>
__global__ __launch_bounds__(256) void gemm_wmma(const void* __restrict__ Aptr,
                                                 const float* __restrict__ Bw,
                                                 void* __restrict__ Cptr,
                                                 int M, int N, int K) {
    __shared__ _Float16 As[2][128][32];   // row-major  [m][k]
    __shared__ _Float16 Bt[2][128][32];   // transposed [n][k]

    const int tid  = threadIdx.x;
    const int wave = tid >> 5;
    const int lane = tid & 31;
    const int rowBase = blockIdx.y * 128;
    const int colBase = blockIdx.x * 128;

    const float*    Af = (const float*)Aptr;
    const _Float16* Ah = (const _Float16*)Aptr;

    v8f acc[8] = {};
    float aR[16], bR[16];

    auto loadRegs = [&](int k0) {
#pragma unroll
        for (int j = 0; j < 16; ++j) {          // A tile 128x32
            int idx = tid + j * 256;
            int r = idx >> 5, c = idx & 31;
            aR[j] = A_HALF ? (float)Ah[(size_t)(rowBase + r) * K + k0 + c]
                           : Af[(size_t)(rowBase + r) * K + k0 + c];
        }
#pragma unroll
        for (int j = 0; j < 16; ++j) {          // B tile 32x128
            int idx = tid + j * 256;
            int kk = idx >> 7, n = idx & 127;
            bR[j] = Bw[(size_t)(k0 + kk) * N + colBase + n];
        }
    };
    auto storeLDS = [&](int buf) {
#pragma unroll
        for (int j = 0; j < 16; ++j) {
            int idx = tid + j * 256;
            int r = idx >> 5, c = idx & 31;
            As[buf][r][c] = (_Float16)aR[j];
        }
#pragma unroll
        for (int j = 0; j < 16; ++j) {
            int idx = tid + j * 256;
            int kk = idx >> 7, n = idx & 127;
            Bt[buf][n][kk] = (_Float16)bR[j];
        }
    };

    loadRegs(0);
    storeLDS(0);
    __syncthreads();

    for (int k0 = 0; k0 < K; k0 += 32) {
        const int  buf  = (k0 >> 5) & 1;
        const bool more = (k0 + 32) < K;
        if (more) loadRegs(k0 + 32);            // global loads for next tile

        v16h afrag = load_A_frag(&As[buf][wave * 16][0], 32, lane);
#pragma unroll
        for (int nt = 0; nt < 8; ++nt) {
            v16h bfrag = load_B_fragT(&Bt[buf][nt * 16][0], 32, lane);
            acc[nt] = __builtin_amdgcn_wmma_f32_16x16x32_f16(
                false, afrag, false, bfrag, (short)0, acc[nt], false, false);
        }

        if (more) storeLDS(buf ^ 1);            // stage next tile into other buffer
        __syncthreads();
    }

    // C/D layout: M = r + 8*(lane>>4), N = lane&15 (+16*nt)
    const int mrow = rowBase + wave * 16 + ((lane >> 4) << 3);
    const int ncol = colBase + (lane & 15);
#pragma unroll
    for (int nt = 0; nt < 8; ++nt)
#pragma unroll
        for (int r = 0; r < 8; ++r) {
            size_t off = (size_t)(mrow + r) * N + ncol + nt * 16;
            if (OUT_FLOAT) ((float*)Cptr)[off] = acc[nt][r];
            else           ((_Float16*)Cptr)[off] = (_Float16)acc[nt][r];
        }
}

// ---------- RoPE (in-place on f16 (B*S, heads*128)) ----------
__global__ __launch_bounds__(256) void rope_kernel(_Float16* t, int heads) {
    size_t idx = (size_t)blockIdx.x * 256 + threadIdx.x;
    size_t total = (size_t)B_ * S_ * heads * (HD_ / 2);
    if (idx >= total) return;
    int    pair  = (int)(idx & 63);
    size_t th    = idx >> 6;
    int    head  = (int)(th % heads);
    size_t token = th / heads;               // b*S + s
    int    s     = (int)(token % S_);

    float inv = __expf(-(2.0f * pair / (float)HD_) * logf(THETA_));
    float ang = (float)s * inv;
    float c, sn;
    sincosf(ang, &c, &sn);

    size_t off = token * ((size_t)heads * HD_) + (size_t)head * HD_ + pair * 2;
    float a = (float)t[off], b = (float)t[off + 1];
    t[off]     = (_Float16)(a * c - b * sn);
    t[off + 1] = (_Float16)(a * sn + b * c);
}

// ---------- Flash attention: grid (S/128, NH, B), 256 threads ----------
__global__ __launch_bounds__(256) void attn_wmma(const _Float16* __restrict__ qh,
                                                 const _Float16* __restrict__ kh,
                                                 const _Float16* __restrict__ vh,
                                                 _Float16* __restrict__ oh) {
    __shared__ _Float16 Ks[64][128];      // [kpos][d]   row-major (K^T B-frags contiguous)
    __shared__ _Float16 Vts[128][64];     // [d][kpos]   transposed (V B-frags contiguous)
    __shared__ _Float16 Ps[8][16][64];    // per-wave P tile, row-major [m][kpos]

    const int qblk = blockIdx.x, h = blockIdx.y, b = blockIdx.z;
    const int kvh  = h >> 2;                       // n_rep = 4
    const int tid  = threadIdx.x;
    const int wave = tid >> 5, lane = tid & 31;
    const float scale = 0.08838834764831845f;      // 1/sqrt(128)

    // Q fragments in registers: 16 rows x 128 dim per wave
    v16h qfrag[4];
    {
        const _Float16* qbase = qh + ((size_t)(b * S_ + qblk * 128 + wave * 16)) * (NH_ * HD_) + h * HD_;
#pragma unroll
        for (int ks = 0; ks < 4; ++ks) qfrag[ks] = load_A_frag(qbase + ks * 32, NH_ * HD_, lane);
    }

    float m_i[8], l_i[8];
    v8f   o[8] = {};
#pragma unroll
    for (int r = 0; r < 8; ++r) { m_i[r] = -1e30f; l_i[r] = 0.0f; }

    const int qrow0 = qblk * 128 + wave * 16 + ((lane >> 4) << 3);
    const int nkb   = 2 * qblk + 2;

    for (int kb = 0; kb < nkb; ++kb) {
        // K tile: async DMA straight into LDS (16B per lane, 1024 chunks)
#pragma unroll
        for (int j = 0; j < 4; ++j) {
            int c  = tid + j * 256;
            int kp = c >> 4;              // 16 x 8-half chunks per 128-half row
            int dc = c & 15;
            const _Float16* g = kh + (size_t)(b * S_ + kb * 64 + kp) * (NKV_ * HD_)
                                   + kvh * HD_ + dc * 8;
            async_copy_b128(g, &Ks[kp][dc * 8]);
        }
        // V tile: contiguous b128 loads into registers, transpose via LDS stores
#pragma unroll
        for (int j = 0; j < 4; ++j) {
            int c  = tid + j * 256;       // 1024 x 8-half chunks
            int kp = c >> 4;
            int dc = c & 15;
            v8h vv = *(const v8h*)(vh + (size_t)(b * S_ + kb * 64 + kp) * (NKV_ * HD_)
                                      + kvh * HD_ + dc * 8);
#pragma unroll
            for (int i = 0; i < 8; ++i) Vts[dc * 8 + i][kp] = vv[i];
        }
        wait_asynccnt0();
        __syncthreads();

        // S = Q @ K^T  (per wave: 16 x 64)
        v8f s[4] = {};
#pragma unroll
        for (int ks = 0; ks < 4; ++ks)
#pragma unroll
            for (int nt = 0; nt < 4; ++nt) {
                v16h bfrag = load_B_fragT(&Ks[nt * 16][ks * 32], 128, lane);
                s[nt] = __builtin_amdgcn_wmma_f32_16x16x32_f16(
                    false, qfrag[ks], false, bfrag, (short)0, s[nt], false, false);
            }

        // scale + causal mask + online softmax
        const int kcol0 = kb * 64 + (lane & 15);
#pragma unroll
        for (int r = 0; r < 8; ++r) {
            float rowmax = -1e30f;
#pragma unroll
            for (int nt = 0; nt < 4; ++nt) {
                float v = s[nt][r] * scale;
                if (kcol0 + nt * 16 > qrow0 + r) v = -1e9f;
                s[nt][r] = v;
                rowmax = fmaxf(rowmax, v);
            }
#pragma unroll
            for (int mk = 1; mk < 16; mk <<= 1)
                rowmax = fmaxf(rowmax, __shfl_xor(rowmax, mk, 32));
            float mnew  = fmaxf(m_i[r], rowmax);
            float alpha = __expf(m_i[r] - mnew);
            m_i[r] = mnew;
            float psum = 0.0f;
#pragma unroll
            for (int nt = 0; nt < 4; ++nt) {
                float p = __expf(s[nt][r] - mnew);
                psum += p;
                Ps[wave][r + ((lane >> 4) << 3)][nt * 16 + (lane & 15)] = (_Float16)p;
            }
            l_i[r] = l_i[r] * alpha + psum;
#pragma unroll
            for (int dt = 0; dt < 8; ++dt) o[dt][r] *= alpha;
        }
        __syncthreads();

        // O += P @ V   (K = 64 -> two k-steps)
#pragma unroll
        for (int ks = 0; ks < 2; ++ks) {
            v16h pfrag = load_A_frag(&Ps[wave][0][ks * 32], 64, lane);
#pragma unroll
            for (int dt = 0; dt < 8; ++dt) {
                v16h vfrag = load_B_fragT(&Vts[dt * 16][ks * 32], 64, lane);
                o[dt] = __builtin_amdgcn_wmma_f32_16x16x32_f16(
                    false, pfrag, false, vfrag, (short)0, o[dt], false, false);
            }
        }
        __syncthreads();
    }

    // finalize: full row sums across the 16-lane group, normalize, store f16
#pragma unroll
    for (int r = 0; r < 8; ++r)
#pragma unroll
        for (int mk = 1; mk < 16; mk <<= 1) l_i[r] += __shfl_xor(l_i[r], mk, 32);

    const size_t trow = (size_t)b * S_ + qblk * 128 + wave * 16 + ((lane >> 4) << 3);
#pragma unroll
    for (int dt = 0; dt < 8; ++dt)
#pragma unroll
        for (int r = 0; r < 8; ++r)
            oh[(trow + r) * (NH_ * HD_) + h * HD_ + dt * 16 + (lane & 15)] =
                (_Float16)(o[dt][r] / l_i[r]);
}

// ---------- host launcher ----------
extern "C" void kernel_launch(void* const* d_in, const int* in_sizes, int n_in,
                              void* d_out, int out_size, void* d_ws, size_t ws_size,
                              hipStream_t stream) {
    const float* x  = (const float*)d_in[0];
    const float* wq = (const float*)d_in[1];
    const float* wk = (const float*)d_in[2];
    const float* wv = (const float*)d_in[3];
    const float* wo = (const float*)d_in[4];
    float* out = (float*)d_out;

    const size_t TOK = (size_t)B_ * S_;            // 4096 tokens
    _Float16* qhp = (_Float16*)d_ws;               // TOK x 4096
    _Float16* khp = qhp + TOK * (NH_ * HD_);       // TOK x 1024
    _Float16* vhp = khp + TOK * (NKV_ * HD_);      // TOK x 1024
    _Float16* ahp = vhp + TOK * (NKV_ * HD_);      // TOK x 4096

    dim3 blk(256);

    // QKV projections (f32 in -> f16 out, f32 accumulation in WMMA)
    gemm_wmma<false, false><<<dim3(NH_ * HD_ / 128, TOK / 128), blk, 0, stream>>>(
        x, wq, qhp, (int)TOK, NH_ * HD_, DIM_);
    gemm_wmma<false, false><<<dim3(NKV_ * HD_ / 128, TOK / 128), blk, 0, stream>>>(
        x, wk, khp, (int)TOK, NKV_ * HD_, DIM_);
    gemm_wmma<false, false><<<dim3(NKV_ * HD_ / 128, TOK / 128), blk, 0, stream>>>(
        x, wv, vhp, (int)TOK, NKV_ * HD_, DIM_);

    // RoPE on q and k
    {
        size_t tq = TOK * NH_ * (HD_ / 2);
        size_t tk = TOK * NKV_ * (HD_ / 2);
        rope_kernel<<<dim3((unsigned)((tq + 255) / 256)), blk, 0, stream>>>(qhp, NH_);
        rope_kernel<<<dim3((unsigned)((tk + 255) / 256)), blk, 0, stream>>>(khp, NKV_);
    }

    // Flash attention
    attn_wmma<<<dim3(S_ / 128, NH_, B_), blk, 0, stream>>>(qhp, khp, vhp, ahp);

    // Output projection (f16 A in -> f32 out)
    gemm_wmma<true, true><<<dim3(DIM_ / 128, TOK / 128), blk, 0, stream>>>(
        ahp, wo, out, (int)TOK, DIM_, DIM_);
}
